// MemoryMultiHeadAttention_19258633355775
// MI455X (gfx1250) — matrix operand order repaired and checked by reading, not compiled, not run on hardware
//
#include <hip/hip_runtime.h>
#include <hip/hip_bf16.h>

// ---------------------------------------------------------------------------
// MemoryMultiHeadAttention on gfx1250 (MI455X): bf16 WMMA + TDM async staging.
// B=2 S=2048 D=1024 H=16 DH=64 M=40 L=2088
// ---------------------------------------------------------------------------

typedef __bf16 bf16_t;
typedef __attribute__((ext_vector_type(16))) __bf16 v16bf;
typedef __attribute__((ext_vector_type(8)))  float  v8f;
typedef __attribute__((ext_vector_type(4)))  unsigned int u32v4;
typedef __attribute__((ext_vector_type(8)))  int i32v8;
typedef __attribute__((ext_vector_type(4)))  int i32v4;

constexpr int B_  = 2;
constexpr int S_  = 2048;
constexpr int D_  = 1024;
constexpr int H_  = 16;
constexpr int M_  = 40;
constexpr int DH_ = 64;
constexpr int L_  = S_ + M_;          // 2088
constexpr int NT_ = (L_ + 15) / 16;   // 131 kv tiles of 16
constexpr int NC_ = (L_ + 31) / 32;   // 66 kv chunks of 32

__device__ inline v8f wmma_bf16(v16bf a, v16bf b, v8f c) {
  return __builtin_amdgcn_wmma_f32_16x16x32_bf16(
      /*neg_a=*/false, a, /*neg_b=*/false, b,
      /*c_mod=*/(short)0, c, /*reuse_a=*/false, /*reuse_b=*/false);
}

// Load a 32-byte fragment from two 16-byte-aligned runs.
__device__ inline v16bf load_frag(const bf16_t* p0, const bf16_t* p1) {
  union { v16bf v; uint4 q[2]; } u;
  u.q[0] = *(const uint4*)(const void*)p0;
  u.q[1] = *(const uint4*)(const void*)p1;
  return u.v;
}

// ---------------------------------------------------------------------------
// TDM: async 2-D tile load global -> LDS.
// Tile: 32 rows x 64 bf16 elems, row stride D_ elems. Rows >= rows_rem are
// zero-filled by TDM OOB handling (tensor_dim1 = rows_rem).
// ---------------------------------------------------------------------------
__device__ inline void tdm_load_tile(unsigned lds_off, const bf16_t* gaddr,
                                     int rows_rem) {
  unsigned long long ga = (unsigned long long)(uintptr_t)gaddr;
  u32v4 g0;
  g0.x = 1u;                                           // count=1 (valid D#)
  g0.y = lds_off;                                      // lds_addr
  g0.z = (unsigned)(ga & 0xffffffffu);                 // global_addr[31:0]
  g0.w = (unsigned)((ga >> 32) & 0x01ffffffu) | (2u << 30);  // addr hi | type=2
  unsigned rem = (unsigned)rows_rem;
  i32v8 g1;
  g1[0] = (int)(1u << 16);                 // workgroup_mask=0, data_size=1 (2B)
  g1[1] = (int)(64u << 16);                // tensor_dim0[15:0] = 64
  g1[2] = (int)((rem & 0xffffu) << 16);    // tensor_dim0 hi=0, tensor_dim1 lo
  g1[3] = (int)((64u << 16) | ((rem >> 16) & 0xffffu)); // tile_dim0=64 | dim1 hi
  g1[4] = 32;                              // tile_dim1=32, tile_dim2=0
  g1[5] = (int)D_;                         // tensor_dim0_stride = 1024
  g1[6] = 0;                               // stride hi / dim1_stride lo
  g1[7] = 0;
  i32v4 gz4 = {0, 0, 0, 0};
  i32v8 gz8 = {0, 0, 0, 0, 0, 0, 0, 0};
  __builtin_amdgcn_tensor_load_to_lds(g0, g1, gz4, gz4, gz8, 0);
}

__device__ inline unsigned lds_off_of(const void* p) {
  return (unsigned)(uintptr_t)p;   // low 32 bits of LDS aperture addr = offset
}

// ---------------------------------------------------------------------------
// f32 -> bf16 conversion
// ---------------------------------------------------------------------------
__global__ void cvt_f32_bf16(const float* __restrict__ in,
                             bf16_t* __restrict__ out, size_t n) {
  size_t i = (size_t)blockIdx.x * blockDim.x + threadIdx.x;
  if (i < n) out[i] = (bf16_t)in[i];
}

// ---------------------------------------------------------------------------
// GEMM: out[orow, :] = A[row, :] @ W + bias     (K = N = 1024)
// orow = (row / rows_per_seg) * seg_stride + (row % rows_per_seg)
// ---------------------------------------------------------------------------
template <bool OUT_F32>
__global__ __launch_bounds__(256) void gemm_bias_kernel(
    const bf16_t* __restrict__ A, const bf16_t* __restrict__ W,
    const float* __restrict__ bias, void* __restrict__ outv,
    int Mrows, int rows_per_seg, int seg_stride) {
  __shared__ __align__(16) bf16_t sA[128 * 32];
  __shared__ __align__(16) bf16_t sB[64 * 32];  // transposed [n][k]

  const int tid  = threadIdx.x;
  const int lane = tid & 31;
  const int wave = tid >> 5;
  const int half = lane >> 4;
  const int l15  = lane & 15;
  const int mblk = blockIdx.y * 128;
  const int nblk = blockIdx.x * 64;
  const int mw   = (wave & 3) * 32;
  const int nw   = (wave >> 2) * 32;

  v8f acc[2][2] = {};

  for (int k0 = 0; k0 < D_; k0 += 32) {
    {
      int r = tid >> 1;
      int c = (tid & 1) * 16;
      int gr = mblk + r;
      uint4 z0 = {0u, 0u, 0u, 0u}, z1 = {0u, 0u, 0u, 0u};
      if (gr < Mrows) {
        const uint4* src = (const uint4*)(const void*)(A + (size_t)gr * D_ + k0 + c);
        z0 = src[0];
        z1 = src[1];
      }
      uint4* dst = (uint4*)(void*)(sA + r * 32 + c);
      dst[0] = z0;
      dst[1] = z1;
    }
    {
      int k  = tid & 31;
      int n0 = (tid >> 5) * 8;
      const bf16_t* src = W + (size_t)(k0 + k) * D_ + nblk + n0;
#pragma unroll
      for (int i = 0; i < 8; ++i) sB[(n0 + i) * 32 + k] = src[i];
    }
    __syncthreads();

#pragma unroll
    for (int i = 0; i < 2; ++i) {
      int arow = mw + i * 16 + l15;
      v16bf a = load_frag(sA + arow * 32 + 8 * half,
                          sA + arow * 32 + 16 + 8 * half);
#pragma unroll
      for (int j = 0; j < 2; ++j) {
        int bcol = nw + j * 16 + l15;
        v16bf b = load_frag(sB + bcol * 32 + 8 * half,
                            sB + bcol * 32 + 16 + 8 * half);
        acc[i][j] = wmma_bf16(a, b, acc[i][j]);
      }
    }
    __syncthreads();
  }

#pragma unroll
  for (int i = 0; i < 2; ++i)
#pragma unroll
    for (int j = 0; j < 2; ++j) {
      int col = nblk + nw + j * 16 + l15;
      float bv = bias ? bias[col] : 0.0f;
#pragma unroll
      for (int r = 0; r < 8; ++r) {
        int row = mblk + mw + i * 16 + half * 8 + r;
        if (row >= Mrows) continue;
        int orow = (row / rows_per_seg) * seg_stride + (row % rows_per_seg);
        float v = acc[i][j][r] + bv;
        if (OUT_F32)
          ((float*)outv)[(size_t)orow * D_ + col] = v;
        else
          ((bf16_t*)outv)[(size_t)orow * D_ + col] = (bf16_t)v;
      }
    }
}

// ---------------------------------------------------------------------------
// Attention. Block 256 = 8 waves, each wave one 16-row q tile (128 q rows).
// K/V kv-chunks (32 rows x 64) staged once per block by wave 0 via TDM,
// double-buffered, shared by all waves through block barriers.
// ---------------------------------------------------------------------------
__global__ __launch_bounds__(256) void attn_kernel(
    const bf16_t* __restrict__ qp, const bf16_t* __restrict__ kc,
    const bf16_t* __restrict__ vc, const float* __restrict__ mask,
    float* __restrict__ attn_out, bf16_t* __restrict__ ctx) {
  __shared__ __align__(16) bf16_t sK[2][32 * 64];   // 8 KB
  __shared__ __align__(16) bf16_t sV[2][32 * 64];   // 8 KB
  __shared__ __align__(16) bf16_t sP[8][16 * 32];   // 8 KB  (per-wave bf16 P)
  __shared__ __align__(16) float  sPF[8][16 * 32];  // 16 KB (per-wave f32 P)

  const int tid  = threadIdx.x;
  const int lane = tid & 31;
  const int wave = tid >> 5;
  const int half = lane >> 4;
  const int l15  = lane & 15;

  const int bid   = blockIdx.x;
  const int qblk  = (bid & 15) * 128;       // S/128 = 16
  const int h     = (bid >> 4) & (H_ - 1);
  const int b     = bid >> 8;
  const int qbase = qblk + wave * 16;

  const size_t hd = (size_t)h * DH_;
  const bf16_t* qrowp = qp + ((size_t)(b * S_ + qbase)) * D_ + hd;
  const bf16_t* kbase = kc + (size_t)b * L_ * D_ + hd;
  const bf16_t* vbase = vc + (size_t)b * L_ * D_ + hd;
  const float*  mrow  = mask + ((size_t)b * S_ + qbase) * L_;
  float* arow = attn_out + ((size_t)(b * H_ + h) * S_ + qbase) * L_;

  // Q fragments for k-dim [0,32) and [32,64)
  v16bf qa[2];
  {
    const bf16_t* p = qrowp + (size_t)l15 * D_;
    qa[0] = load_frag(p + 8 * half, p + 16 + 8 * half);
    qa[1] = load_frag(p + 32 + 8 * half, p + 48 + 8 * half);
  }

  // scores for 16-wide sub-tile t of the staged 32-row chunk (branch-free)
  auto score_tile = [&](const bf16_t* Kb, int t, int lt, v8f& s) {
    const bf16_t* kp = Kb + (t * 16 + l15) * 64;
    v16bf b0 = load_frag(kp + 8 * half, kp + 16 + 8 * half);
    v16bf b1 = load_frag(kp + 32 + 8 * half, kp + 48 + 8 * half);
    s = {};
    s = wmma_bf16(qa[0], b0, s);
    s = wmma_bf16(qa[1], b1, s);
    int kv  = lt * 16 + l15;
    int kvc = kv < L_ ? kv : (L_ - 1);
    float pad = (kv < L_) ? 0.0f : -3.0e38f;
#pragma unroll
    for (int r = 0; r < 8; ++r) {
      int qr = half * 8 + r;
      float mv = mrow[(size_t)qr * L_ + kvc];
      s[r] = s[r] * 0.125f + mv * (-1e9f) + pad;
    }
  };

  // ---- Pass 1: streaming row max / sum --------------------------------
  float m[8], l[8];
#pragma unroll
  for (int r = 0; r < 8; ++r) { m[r] = -1e30f; l[r] = 0.0f; }

  if (wave == 0) tdm_load_tile(lds_off_of(&sK[0][0]), kbase, L_);

  for (int c = 0; c < NC_; ++c) {
    if (wave == 0) {
      if (c + 1 < NC_) {
        tdm_load_tile(lds_off_of(&sK[(c + 1) & 1][0]),
                      kbase + (size_t)(c + 1) * 32 * D_, L_ - (c + 1) * 32);
        __builtin_amdgcn_s_wait_tensorcnt(1);
      } else {
        __builtin_amdgcn_s_wait_tensorcnt(0);
      }
    }
    __syncthreads();
    const bf16_t* Kb = sK[c & 1];
#pragma unroll
    for (int t = 0; t < 2; ++t) {
      int lt = 2 * c + t;
      if (lt >= NT_) break;  // uniform: only last chunk's second sub-tile
      v8f s;
      score_tile(Kb, t, lt, s);
#pragma unroll
      for (int r = 0; r < 8; ++r) {
        float v  = s[r];
        float mn = fmaxf(m[r], v);
        l[r] = l[r] * __expf(m[r] - mn) + __expf(v - mn);
        m[r] = mn;
      }
    }
    __syncthreads();
  }
  // merge across the 16 lanes that share the same rows
#pragma unroll
  for (int xm = 1; xm < 16; xm <<= 1) {
#pragma unroll
    for (int r = 0; r < 8; ++r) {
      float mo = __shfl_xor(m[r], xm, 32);
      float lo = __shfl_xor(l[r], xm, 32);
      float mn = fmaxf(m[r], mo);
      l[r] = l[r] * __expf(m[r] - mn) + lo * __expf(mo - mn);
      m[r] = mn;
    }
  }
  float rl[8];
#pragma unroll
  for (int r = 0; r < 8; ++r) rl[r] = (l[r] > 0.0f) ? 1.0f / l[r] : 0.0f;

  // ---- Pass 2: attn write + ctx accumulate ----------------------------
  v8f cacc[4] = {};
  bf16_t* Pb = sP[wave];
  float*  Pf = sPF[wave];

  if (wave == 0) {
    tdm_load_tile(lds_off_of(&sK[0][0]), kbase, L_);
    tdm_load_tile(lds_off_of(&sV[0][0]), vbase, L_);
  }

  for (int c = 0; c < NC_; ++c) {
    if (wave == 0) {
      if (c + 1 < NC_) {
        int rem = L_ - (c + 1) * 32;
        tdm_load_tile(lds_off_of(&sK[(c + 1) & 1][0]),
                      kbase + (size_t)(c + 1) * 32 * D_, rem);
        tdm_load_tile(lds_off_of(&sV[(c + 1) & 1][0]),
                      vbase + (size_t)(c + 1) * 32 * D_, rem);
        __builtin_amdgcn_s_wait_tensorcnt(2);
      } else {
        __builtin_amdgcn_s_wait_tensorcnt(0);
      }
    }
    __syncthreads();
    const bf16_t* Kb = sK[c & 1];
    const bf16_t* Vb = sV[c & 1];

#pragma unroll
    for (int t = 0; t < 2; ++t) {
      int lt = 2 * c + t;
      float p[8];
      if (lt < NT_) {
        v8f s;
        score_tile(Kb, t, lt, s);
#pragma unroll
        for (int r = 0; r < 8; ++r) p[r] = __expf(s[r] - m[r]) * rl[r];
      } else {
#pragma unroll
        for (int r = 0; r < 8; ++r) p[r] = 0.0f;
      }
#pragma unroll
      for (int r = 0; r < 8; ++r) {
        int qr = half * 8 + r;
        Pf[qr * 32 + t * 16 + l15] = p[r];
        Pb[qr * 32 + t * 16 + l15] = (bf16_t)p[r];
      }
    }
    asm volatile("s_wait_dscnt 0" ::: "memory");

    // coalesced attn store: 16 rows x 32 cols f32, 16B per store
#pragma unroll
    for (int j = 0; j < 4; ++j) {
      int idx = lane + 32 * j;   // 0..127
      int row = idx >> 3;
      int seg = idx & 7;
      int col4 = c * 32 + seg * 4;
      if (col4 + 4 <= L_) {
        uint4 val = *(const uint4*)(const void*)(Pf + row * 32 + seg * 4);
        *(uint4*)(void*)(arow + (size_t)row * L_ + col4) = val;
      }
    }

    // P fragment (A layout) + 4 V fragments (strided gather from row-major Vb)
    {
      v16bf pa = load_frag(Pb + l15 * 32 + 8 * half,
                           Pb + l15 * 32 + 16 + 8 * half);
#pragma unroll
      for (int nb = 0; nb < 4; ++nb) {
        union { v16bf v; bf16_t e[16]; } u;
        int dh = nb * 16 + l15;
#pragma unroll
        for (int kk = 0; kk < 16; ++kk)
          u.e[kk] = Vb[(8 * half + (kk & 7) + (kk >> 3) * 16) * 64 + dh];
        cacc[nb] = wmma_bf16(pa, u.v, cacc[nb]);
      }
    }
    __syncthreads();
  }

  // store ctx (bf16) in [B*S, D] layout with head offset
  bf16_t* cp = ctx + ((size_t)(b * S_) + qbase) * D_ + hd;
#pragma unroll
  for (int nb = 0; nb < 4; ++nb)
#pragma unroll
    for (int r = 0; r < 8; ++r) {
      int qr = half * 8 + r;
      cp[(size_t)qr * D_ + nb * 16 + l15] = (bf16_t)cacc[nb][r];
    }
}

// ---------------------------------------------------------------------------
// Host-side launcher
// ---------------------------------------------------------------------------
static inline size_t align256(size_t x) { return (x + 255) & ~(size_t)255; }

extern "C" void kernel_launch(void* const* d_in, const int* in_sizes, int n_in,
                              void* d_out, int out_size, void* d_ws, size_t ws_size,
                              hipStream_t stream) {
  (void)in_sizes; (void)n_in; (void)out_size; (void)ws_size;

  const float* v_f    = (const float*)d_in[0];
  const float* k_f    = (const float*)d_in[1];
  const float* q_f    = (const float*)d_in[2];
  const float* mask   = (const float*)d_in[3];
  const float* Wq_f   = (const float*)d_in[4];
  const float* bq     = (const float*)d_in[5];
  const float* Wk_f   = (const float*)d_in[6];
  const float* bk     = (const float*)d_in[7];
  const float* Wv_f   = (const float*)d_in[8];
  const float* bv     = (const float*)d_in[9];
  const float* kmem_f = (const float*)d_in[10];
  const float* vmem_f = (const float*)d_in[11];
  const float* Wkm_f  = (const float*)d_in[12];
  const float* bkm    = (const float*)d_in[13];
  const float* Wvm_f  = (const float*)d_in[14];
  const float* bvm    = (const float*)d_in[15];
  const float* Wo_f   = (const float*)d_in[16];
  const float* bo     = (const float*)d_in[17];

  const size_t nBSD = (size_t)B_ * S_ * D_;
  const size_t nMD  = (size_t)M_ * D_;
  const size_t nDD  = (size_t)D_ * D_;
  const size_t nBLD = (size_t)B_ * L_ * D_;

  char* ws = (char*)d_ws;
  size_t off = 0;
  auto alloc = [&](size_t bytes) { char* p = ws + off; off += align256(bytes); return p; };

  bf16_t* qb   = (bf16_t*)alloc(nBSD * 2);
  bf16_t* kb   = (bf16_t*)alloc(nBSD * 2);
  bf16_t* vb   = (bf16_t*)alloc(nBSD * 2);
  bf16_t* kmb  = (bf16_t*)alloc(nMD * 2);
  bf16_t* vmb  = (bf16_t*)alloc(nMD * 2);
  bf16_t* Wqb  = (bf16_t*)alloc(nDD * 2);
  bf16_t* Wkb  = (bf16_t*)alloc(nDD * 2);
  bf16_t* Wvb  = (bf16_t*)alloc(nDD * 2);
  bf16_t* Wkmb = (bf16_t*)alloc(nDD * 2);
  bf16_t* Wvmb = (bf16_t*)alloc(nDD * 2);
  bf16_t* Wob  = (bf16_t*)alloc(nDD * 2);
  bf16_t* qpB  = (bf16_t*)alloc(nBSD * 2);
  bf16_t* kcB  = (bf16_t*)alloc(nBLD * 2);
  bf16_t* vcB  = (bf16_t*)alloc(nBLD * 2);
  bf16_t* ctxB = (bf16_t*)alloc(nBSD * 2);

  float* out_f  = (float*)d_out;        // [B,S,D]
  float* attn_f = out_f + nBSD;         // [B,H,S,L]

  auto cvt = [&](const float* src, bf16_t* dst, size_t n) {
    int grid = (int)((n + 255) / 256);
    cvt_f32_bf16<<<grid, 256, 0, stream>>>(src, dst, n);
  };
  cvt(q_f, qb, nBSD);
  cvt(k_f, kb, nBSD);
  cvt(v_f, vb, nBSD);
  cvt(kmem_f, kmb, nMD);
  cvt(vmem_f, vmb, nMD);
  cvt(Wq_f, Wqb, nDD);
  cvt(Wk_f, Wkb, nDD);
  cvt(Wv_f, Wvb, nDD);
  cvt(Wkm_f, Wkmb, nDD);
  cvt(Wvm_f, Wvmb, nDD);
  cvt(Wo_f, Wob, nDD);

  dim3 blk(256);
  dim3 gBig(D_ / 64, (B_ * S_ + 127) / 128);   // 16 x 32
  dim3 gMem(D_ / 64, (M_ + 127) / 128);        // 16 x 1

  gemm_bias_kernel<false><<<gBig, blk, 0, stream>>>(
      qb, Wqb, bq, qpB, B_ * S_, B_ * S_, 0);
  gemm_bias_kernel<false><<<gBig, blk, 0, stream>>>(
      kb, Wkb, bk, kcB, B_ * S_, S_, L_);
  gemm_bias_kernel<false><<<gBig, blk, 0, stream>>>(
      vb, Wvb, bv, vcB, B_ * S_, S_, L_);
  for (int b = 0; b < B_; ++b) {
    gemm_bias_kernel<false><<<gMem, blk, 0, stream>>>(
        kmb, Wkmb, bkm, kcB + ((size_t)b * L_ + S_) * D_, M_, M_, 0);
    gemm_bias_kernel<false><<<gMem, blk, 0, stream>>>(
        vmb, Wvmb, bvm, vcB + ((size_t)b * L_ + S_) * D_, M_, M_, 0);
  }

  attn_kernel<<<dim3(B_ * H_ * (S_ / 128)), blk, 0, stream>>>(
      qpB, kcB, vcB, mask, attn_f, ctxB);

  gemm_bias_kernel<true><<<gBig, blk, 0, stream>>>(
      ctxB, Wob, bo, out_f, B_ * S_, B_ * S_, 0);
}
